// SoftGBM_2430951489981
// MI455X (gfx1250) — matrix operand order repaired and checked by reading, not compiled, not run on hardware
//
#include <hip/hip_runtime.h>

typedef __attribute__((ext_vector_type(16))) _Float16 v16h;
typedef __attribute__((ext_vector_type(8)))  _Float16 v8h;
typedef __attribute__((ext_vector_type(4)))  _Float16 v4h;
typedef __attribute__((ext_vector_type(8)))  float    v8f;

#define TREES 32
#define NNODE 63
#define NLEAF 64
#define KDIM  256
#define PDIM  8
#define MTILE 32
#define XSTR  264   // halves per LDS row (256 + 8 pad -> 16B-aligned, conflict-free)
#define SSTR  68    // floats per s-buffer row (64 + 4 pad)

// ---------------------------------------------------------------------------
// Pre-kernel: convert W (T,63,256) f32 -> f16 into d_ws as (T,64,256), with
// row 63 of every tree zero-filled (padding column for the 16-wide WMMA tiles).
// ---------------------------------------------------------------------------
__global__ __launch_bounds__(256)
void convert_w_kernel(const float* __restrict__ W, _Float16* __restrict__ w16)
{
    int i  = blockIdx.x * 256 + threadIdx.x;   // 512 blocks * 256 = 131072 quads
    int t  = i >> 12;                          // 4096 quads per tree
    int r  = (i >> 6) & 63;                    // row 0..63
    int c4 = (i & 63) << 2;                    // column quad
    v4h h;
    if (r < NNODE) {
        const float4 v = *(const float4*)(W + ((size_t)t * NNODE + r) * KDIM + c4);
        h = (v4h){(_Float16)v.x, (_Float16)v.y, (_Float16)v.z, (_Float16)v.w};
    } else {
        h = (v4h){(_Float16)0.f, (_Float16)0.f, (_Float16)0.f, (_Float16)0.f};
    }
    *(v4h*)(w16 + ((size_t)t * NLEAF + r) * KDIM + c4) = h;
}

// ---------------------------------------------------------------------------
// Main fused kernel
// ---------------------------------------------------------------------------
__global__ __launch_bounds__(256)
void softgbm_wmma_kernel(const float* __restrict__ x,
                         const _Float16* __restrict__ w16,
                         const float* __restrict__ bias,
                         const float* __restrict__ phi,
                         float* __restrict__ out)
{
    __shared__ _Float16 lds_x[MTILE * XSTR];      // 16,896 B  x tile (f16)
    __shared__ _Float16 lds_w[2][NLEAF * XSTR];   // 67,584 B  W[t] ping-pong (f16)
    __shared__ float    lds_s[MTILE * SSTR];      //  8,704 B  sigmoid buf / reduce
    __shared__ float    lds_phi[NLEAF * PDIM];    //  2,048 B  phi[t]

    const int tid   = threadIdx.x;
    const int mbase = blockIdx.x * MTILE;

    // ---------------- stage x tile: f32 -> f16 (once per block) ------------
    for (int i = tid; i < MTILE * (KDIM / 4); i += 256) {
        int r  = i >> 6;
        int c4 = (i & 63) << 2;
        const float4 v = *(const float4*)(x + (size_t)(mbase + r) * KDIM + c4);
        int o = r * XSTR + c4;
        lds_x[o + 0] = (_Float16)v.x;
        lds_x[o + 1] = (_Float16)v.y;
        lds_x[o + 2] = (_Float16)v.z;
        lds_x[o + 3] = (_Float16)v.w;
    }

    // -------- async W-tile stage: 2048 x b128 = 8 instructions/thread ------
    auto issue_w = [&](int t, int sel) {
        const _Float16* gbase = w16 + (size_t)t * (NLEAF * KDIM);
#pragma unroll
        for (int ii = 0; ii < 8; ++ii) {
            int i = ii * 256 + tid;
            int n = i >> 5;              // row 0..63 (one row per wave-issue)
            int c = (i & 31) << 3;       // 16-byte chunk within row (halves)
            unsigned loff =
                (unsigned)(size_t)&lds_w[sel][n * XSTR + c];            // LDS byte offset
            unsigned long long gaddr =
                (unsigned long long)(size_t)(gbase + n * KDIM + c);     // global addr
            asm volatile("global_load_async_to_lds_b128 %0, %1, off"
                         :: "v"(loff), "v"(gaddr) : "memory");
        }
    };

    // wave/lane decomposition for the GEMM stage
    const int lane = tid & 31;
    const int wv   = tid >> 5;   // 8 waves
    const int mh   = wv & 1;     // which 16-row half of the 32-row tile
    const int jt   = wv >> 1;    // which 16-wide column tile (N = 64 total)
    const int lr   = lane & 15;
    const int lh   = lane >> 4;
    const int arow = 16 * mh + lr;   // A fragment: lane -> matrix row
    const int bcol = 16 * jt + lr;   // B fragment: lane -> matrix column

    // leaf-stage decomposition: 32 rows x 8 leaf-groups (8 leaves each)
    const int lm = tid & 31;
    const int lg = tid >> 5;

    float acc[PDIM];
#pragma unroll
    for (int q = 0; q < PDIM; ++q) acc[q] = 0.0f;

    issue_w(0, 0);   // prime the pipeline

    for (int t = 0; t < TREES; ++t) {
        if (t + 1 < TREES) issue_w(t + 1, (t + 1) & 1);

        // stage phi[t] (tiny) while the async copies are in flight
        for (int i = tid; i < NLEAF * PDIM; i += 256)
            lds_phi[i] = phi[(size_t)t * NLEAF * PDIM + i];

        // tree t's 8 async b128s have landed once <=8 remain (in-order completion)
        if (t + 1 < TREES) asm volatile("s_wait_asynccnt 0x8" ::: "memory");
        else               asm volatile("s_wait_asynccnt 0x0" ::: "memory");
        __syncthreads();

        const _Float16* wb = lds_w[t & 1];

        // ---- WMMA GEMM: one 16x16 f32 logit tile per wave, K = 256 --------
        v8f c = {0.f, 0.f, 0.f, 0.f, 0.f, 0.f, 0.f, 0.f};
#pragma unroll
        for (int k = 0; k < KDIM; k += 32) {
            // A (16x32 f16): lane holds row arow, K chunks {ka..ka+7, ka+16..ka+23}
            const int ka = k + 8 * lh;
            v8h a0 = *(const v8h*)(&lds_x[arow * XSTR + ka]);
            v8h a1 = *(const v8h*)(&lds_x[arow * XSTR + ka + 16]);
            v16h a = __builtin_shufflevector(a0, a1,
                       0,1,2,3,4,5,6,7,8,9,10,11,12,13,14,15);
            // B (32x16 f16): lane holds column bcol, K chunks {kb..kb+7, kb+16..kb+23}
            const int kb = k + 8 * lh;
            v8h b0 = *(const v8h*)(&wb[bcol * XSTR + kb]);
            v8h b1 = *(const v8h*)(&wb[bcol * XSTR + kb + 16]);
            v16h bm = __builtin_shufflevector(b0, b1,
                        0,1,2,3,4,5,6,7,8,9,10,11,12,13,14,15);
            c = __builtin_amdgcn_wmma_f32_16x16x32_f16(
                    false, a, false, bm, (short)0, c, false, false);
        }

        // ---- bias + sigmoid -> lds_s (C layout: VGPR r, lane -> M = r + 8*lh)
        const int   ncol = 16 * jt + lr;
        const float bv   = (ncol < NNODE) ? bias[t * NNODE + ncol] : 0.0f;
#pragma unroll
        for (int r = 0; r < 8; ++r) {
            int   row = 16 * mh + 8 * lh + r;
            float z   = c[r] + bv;
            float s   = 1.0f / (1.0f + __expf(-z));
            lds_s[row * SSTR + ncol] = s;
        }
        __syncthreads();

        // ---- leaf probabilities (6-level product) x phi[t], accumulate ----
        const float* srow = &lds_s[lm * SSTR];
#pragma unroll
        for (int li = 0; li < 8; ++li) {
            int   leaf = lg * 8 + li;
            float p    = 1.0f;
#pragma unroll
            for (int d = 0; d < 6; ++d) {
                int   idx = (1 << d) - 1 + (leaf >> (6 - d));
                float sv  = srow[idx];
                p *= ((leaf >> (5 - d)) & 1) ? sv : (1.0f - sv);
            }
#pragma unroll
            for (int q = 0; q < PDIM; ++q)
                acc[q] += p * lds_phi[leaf * PDIM + q];
        }
        __syncthreads();   // protect lds_s/lds_phi and ping-pong buffer reuse
    }

    // ---- reduce 8 leaf-groups per row, apply shrinkage, write output ------
    float* red = lds_s;  // reuse: need 2048 floats, have 2176
#pragma unroll
    for (int q = 0; q < PDIM; ++q)
        red[(lg * MTILE + lm) * PDIM + q] = acc[q];
    __syncthreads();

    const int om = tid >> 3;   // row 0..31
    const int oq = tid & 7;    // p   0..7
    float sum = 0.0f;
#pragma unroll
    for (int g = 0; g < 8; ++g)
        sum += red[(g * MTILE + om) * PDIM + oq];
    out[(size_t)(mbase + om) * PDIM + oq] = 0.1f * sum;
}

extern "C" void kernel_launch(void* const* d_in, const int* in_sizes, int n_in,
                              void* d_out, int out_size, void* d_ws, size_t ws_size,
                              hipStream_t stream)
{
    const float* x    = (const float*)d_in[0];   // (8192, 256)
    const float* W    = (const float*)d_in[1];   // (32, 63, 256)
    const float* bias = (const float*)d_in[2];   // (32, 63)
    const float* phi  = (const float*)d_in[3];   // (32, 64, 8)
    float*       out  = (float*)d_out;           // (8192, 8)
    _Float16*    w16  = (_Float16*)d_ws;         // (32, 64, 256) f16, 1 MB

    (void)in_sizes; (void)n_in; (void)out_size; (void)ws_size;

    convert_w_kernel<<<dim3(512), dim3(256), 0, stream>>>(W, w16);
    softgbm_wmma_kernel<<<dim3(8192 / MTILE), dim3(256), 0, stream>>>(
        x, w16, bias, phi, out);
}
